// GRURegressor_66391604461962
// MI455X (gfx1250) — matrix operand order repaired and checked by reading, not compiled, not run on hardware
//
#include <hip/hip_runtime.h>

// GRU regressor: B=256, T=512, F=64, H=512.
// Persistent-grid bf16 WMMA recurrence:
//  - W_hh gate slices resident in LDS for all 512 steps
//  - per-step h tile staged into LDS with global_load_async_to_lds_b128 (ASYNCcnt)
//  - f32 hidden-state carry + bf16 shadow feeding the WMMA A operand
//  - atomic-counter device barrier between timesteps

#define B_  256
#define T_  512
#define F_  64
#define H_  512
#define NWG 64   // 4 batch tiles (64 rows) x 16 H-slices (32 cols)

typedef __bf16 v16bf __attribute__((ext_vector_type(16)));
typedef float  v8f   __attribute__((ext_vector_type(8)));

// LDS layout (dynamic, byte offsets):
//   Ht  [64][512]  bf16   65536 B   (per-step h tile, async-staged)
//   Wh  [96][512]  bf16   98304 B   (W_hh gate slices, resident)
//   Wi  [96][64]   bf16   12288 B   (W_ih gate slices, resident)
//   GH  [64][96]   f32    24576 B
//   GI  [64][96]   f32    24576 B
#define LDS_BYTES (64 * 512 * 2 + 96 * 512 * 2 + 96 * 64 * 2 + 2 * 64 * 96 * 4)

__device__ __forceinline__ unsigned short f2bf(float f) {
  unsigned u = __float_as_uint(f);
  u += 0x7FFFu + ((u >> 16) & 1u);  // round-to-nearest-even
  return (unsigned short)(u >> 16);
}

// Build a 16-element bf16 fragment from two 16-byte chunks.
__device__ __forceinline__ v16bf frag_ld2(const unsigned short* p0,
                                          const unsigned short* p1) {
  v16bf v;
  ((uint4*)&v)[0] = *(const uint4*)p0;
  ((uint4*)&v)[1] = *(const uint4*)p1;
  return v;
}

__global__ void gru_init_kernel(const float* __restrict__ x,
                                const float* __restrict__ wih,
                                const float* __restrict__ whh,
                                unsigned short* __restrict__ x_bf,
                                unsigned short* __restrict__ wih_bf,
                                unsigned short* __restrict__ whh_bf,
                                unsigned short* __restrict__ h_bf,
                                float* __restrict__ h_f,
                                unsigned* __restrict__ cnt) {
  size_t tid = (size_t)blockIdx.x * blockDim.x + threadIdx.x;
  size_t n   = (size_t)gridDim.x * blockDim.x;
  for (size_t i = tid; i < (size_t)B_ * T_ * F_; i += n) x_bf[i]   = f2bf(x[i]);
  for (size_t i = tid; i < (size_t)3 * H_ * F_; i += n) wih_bf[i] = f2bf(wih[i]);
  for (size_t i = tid; i < (size_t)3 * H_ * H_; i += n) whh_bf[i] = f2bf(whh[i]);
  for (size_t i = tid; i < (size_t)B_ * H_; i += n) { h_bf[i] = 0; h_f[i] = 0.0f; }
  if (tid == 0) *cnt = 0u;
}

__global__ void __launch_bounds__(256) gru_persist_kernel(
    const unsigned short* __restrict__ x_bf,    // (B,T,F) bf16
    const unsigned short* __restrict__ wih_bf,  // (3H,F)  bf16, rows [r;z;n]
    const unsigned short* __restrict__ whh_bf,  // (3H,H)  bf16
    const float* __restrict__ b_ih,
    const float* __restrict__ b_hh,
    unsigned short* __restrict__ h_bf,          // (B,H) bf16 shadow of h
    float* __restrict__ h_f,                    // (B,H) f32 carry
    unsigned* __restrict__ cnt) {
  extern __shared__ unsigned short smem[];
  unsigned short* Ht = smem;                 // [64][512] bf16 (h tile)
  unsigned short* Wh = Ht + 64 * 512;        // [96][512] bf16
  unsigned short* Wi = Wh + 96 * 512;        // [96][64]  bf16
  float* GH = (float*)(Wi + 96 * 64);        // [64][96] hidden-proj tile
  float* GI = GH + 64 * 96;                  // [64][96] input-proj tile

  const int tid  = threadIdx.x;
  const int lane = tid & 31;
  const int wave = tid >> 5;
  const int b0 = (blockIdx.x & 3) * 64;      // batch tile base
  const int j0 = (blockIdx.x >> 2) * 32;     // H-column slice base
  const int m0 = (wave & 3) * 16;            // wave's 16-row sub-tile
  const int nb = (wave >> 2) * 48;           // wave's local column base (3 tiles)
  const int arow = lane & 15;
  const int kA = (lane >> 4) << 3;           // A K-chunk offset: 0 or 8
  const int kB = (lane >> 4) << 4;           // B K-chunk offset: 0 or 16

  // Wave-relative LDS byte address of the h tile (flat->LDS keeps addr[31:0]).
  const unsigned ht_lds_base = (unsigned)(uintptr_t)(void*)Ht;

  // Stage this WG's W_hh / W_ih gate slices into LDS once (local row =
  // gate*32 + jl, gate 0=reset 1=update 2=new; global row = gate*H + j0 + jl).
  for (int idx = tid; idx < 96 * 64; idx += 256) {
    int r = idx >> 6, c = (idx & 63) << 3;
    int g = r >> 5, jl = r & 31;
    *(uint4*)&Wh[r * 512 + c] =
        *(const uint4*)(whh_bf + (size_t)(g * H_ + j0 + jl) * H_ + c);
  }
  for (int idx = tid; idx < 96 * 8; idx += 256) {
    int r = idx >> 3, c = (idx & 7) << 3;
    int g = r >> 5, jl = r & 31;
    *(uint4*)&Wi[r * 64 + c] =
        *(const uint4*)(wih_bf + (size_t)(g * H_ + j0 + jl) * F_ + c);
  }
  __syncthreads();

  const unsigned short* Brow0  = Wh + (size_t)(nb + 0  + arow) * 512;
  const unsigned short* Brow1  = Wh + (size_t)(nb + 16 + arow) * 512;
  const unsigned short* Brow2  = Wh + (size_t)(nb + 32 + arow) * 512;
  const unsigned short* BIrow0 = Wi + (size_t)(nb + 0  + arow) * 64;
  const unsigned short* BIrow1 = Wi + (size_t)(nb + 16 + arow) * 64;
  const unsigned short* BIrow2 = Wi + (size_t)(nb + 32 + arow) * 64;
  const unsigned short* AhL    = Ht + (size_t)(m0 + arow) * 512;  // A rows in LDS

  for (int t = 0; t < T_; ++t) {
    // ---- Async-stage the 64x512 bf16 h tile into LDS (64 KB = 4096 x b128).
    // Each thread fires 16 global_load_async_to_lds_b128 (tracked by ASYNCcnt).
    #pragma unroll
    for (int i = 0; i < 16; ++i) {
      int idx = i * 256 + tid;            // 0..4095 16-byte chunks
      int row = idx >> 6;                 // 64 chunks per 512-elem row
      int c16 = (idx & 63) << 4;          // byte offset within row
      unsigned ldsoff = ht_lds_base + (unsigned)(row * 1024 + c16);
      unsigned long long ga =
          (unsigned long long)(const void*)(h_bf + (size_t)(b0 + row) * H_) +
          (unsigned)c16;
      asm volatile("global_load_async_to_lds_b128 %0, %1, off"
                   :: "v"(ldsoff), "v"(ga) : "memory");
    }
    asm volatile("s_wait_asynccnt 0" ::: "memory");
    __syncthreads();  // all waves' async LDS writes visible

    v8f gi0 = {}, gi1 = {}, gi2 = {};
    v8f gh0 = {}, gh1 = {}, gh2 = {};

    const unsigned short* Axrow =
        x_bf + ((size_t)(b0 + m0 + arow) * T_ + t) * F_;
    __builtin_prefetch(Axrow + F_, 0, 1);  // next timestep's x row

    // Input projection: K = 64 (2 WMMA k-steps), x_t @ W_ih^T.
    #pragma unroll
    for (int kk = 0; kk < F_; kk += 32) {
      v16bf a  = frag_ld2(Axrow + kk + kB, Axrow + kk + kB + 8);
      v16bf w0 = frag_ld2(BIrow0 + kk + kB, BIrow0 + kk + kB + 8);
      v16bf w1 = frag_ld2(BIrow1 + kk + kB, BIrow1 + kk + kB + 8);
      v16bf w2 = frag_ld2(BIrow2 + kk + kB, BIrow2 + kk + kB + 8);
      gi0 = __builtin_amdgcn_wmma_f32_16x16x32_bf16(false, a, false, w0, (short)0, gi0, false, false);
      gi1 = __builtin_amdgcn_wmma_f32_16x16x32_bf16(false, a, false, w1, (short)0, gi1, false, false);
      gi2 = __builtin_amdgcn_wmma_f32_16x16x32_bf16(false, a, false, w2, (short)0, gi2, false, false);
    }

    // Hidden projection: K = 512 (16 WMMA k-steps), A and B both from LDS.
    #pragma unroll 2
    for (int kk = 0; kk < H_; kk += 32) {
      v16bf a  = frag_ld2(AhL + kk + kB, AhL + kk + kB + 8);
      v16bf w0 = frag_ld2(Brow0 + kk + kB, Brow0 + kk + kB + 8);
      v16bf w1 = frag_ld2(Brow1 + kk + kB, Brow1 + kk + kB + 8);
      v16bf w2 = frag_ld2(Brow2 + kk + kB, Brow2 + kk + kB + 8);
      gh0 = __builtin_amdgcn_wmma_f32_16x16x32_bf16(false, a, false, w0, (short)0, gh0, false, false);
      gh1 = __builtin_amdgcn_wmma_f32_16x16x32_bf16(false, a, false, w1, (short)0, gh1, false, false);
      gh2 = __builtin_amdgcn_wmma_f32_16x16x32_bf16(false, a, false, w2, (short)0, gh2, false, false);
    }

    __syncthreads();  // done reading Ht/GH/GI from previous phases

    // Scatter accumulators to LDS. C/D layout: VGPR i -> M = i + 8*(lane>=16),
    // N = lane & 15.
    #pragma unroll
    for (int i = 0; i < 8; ++i) {
      int lm = (lane < 16) ? i : (8 + i);
      int ln = lane & 15;
      float* gh = &GH[(m0 + lm) * 96 + nb + ln];
      float* gi = &GI[(m0 + lm) * 96 + nb + ln];
      gh[0] = gh0[i]; gh[16] = gh1[i]; gh[32] = gh2[i];
      gi[0] = gi0[i]; gi[16] = gi1[i]; gi[32] = gi2[i];
    }
    __syncthreads();

    // Fused GRU gating in f32: 64x32 h_new elements, 8 per thread.
    #pragma unroll
    for (int e = 0; e < 8; ++e) {
      int id = e * 256 + tid;
      int bl = id >> 5;
      int jl = id & 31;
      int j  = j0 + jl;
      const float* gh = &GH[bl * 96 + jl];
      const float* gi = &GI[bl * 96 + jl];
      float r = 1.0f / (1.0f + __expf(-(gi[0]  + b_ih[j]          + gh[0]  + b_hh[j])));
      float z = 1.0f / (1.0f + __expf(-(gi[32] + b_ih[H_ + j]     + gh[32] + b_hh[H_ + j])));
      float n = tanhf(gi[64] + b_ih[2 * H_ + j] + r * (gh[64] + b_hh[2 * H_ + j]));
      size_t hidx = (size_t)(b0 + bl) * H_ + j;
      float hn = (1.0f - z) * n + z * h_f[hidx];
      h_f[hidx]  = hn;
      h_bf[hidx] = f2bf(hn);
    }

    // Device-wide step barrier (64 persistent WGs).
    __syncthreads();
    if (tid == 0) {
      __threadfence();
      atomicAdd(cnt, 1u);
      const unsigned tgt = (unsigned)NWG * (unsigned)(t + 1);
      while (__hip_atomic_load(cnt, __ATOMIC_RELAXED, __HIP_MEMORY_SCOPE_AGENT) < tgt)
        __builtin_amdgcn_s_sleep(2);
      __threadfence();
    }
    __syncthreads();
  }
}

__global__ void gru_head_kernel(const float* __restrict__ h_f,
                                const float* __restrict__ head_w,
                                const float* __restrict__ head_b,
                                float* __restrict__ y) {
  int b = blockIdx.x * blockDim.x + threadIdx.x;
  if (b < B_) {
    float s = head_b[0];
    const float* hr = h_f + (size_t)b * H_;
    for (int k = 0; k < H_; ++k) s += hr[k] * head_w[k];
    y[b] = s;
  }
}

extern "C" void kernel_launch(void* const* d_in, const int* in_sizes, int n_in,
                              void* d_out, int out_size, void* d_ws, size_t ws_size,
                              hipStream_t stream) {
  (void)in_sizes; (void)n_in; (void)out_size; (void)ws_size;
  const float* x     = (const float*)d_in[0];
  const float* W_ih  = (const float*)d_in[1];
  const float* W_hh  = (const float*)d_in[2];
  const float* b_ih  = (const float*)d_in[3];
  const float* b_hh  = (const float*)d_in[4];
  const float* headw = (const float*)d_in[5];
  const float* headb = (const float*)d_in[6];
  float* y = (float*)d_out;

  // Workspace layout (~19.3 MB total).
  char* p = (char*)d_ws;
  unsigned short* x_bf   = (unsigned short*)p; p += (size_t)B_ * T_ * F_ * 2;
  unsigned short* wih_bf = (unsigned short*)p; p += (size_t)3 * H_ * F_ * 2;
  unsigned short* whh_bf = (unsigned short*)p; p += (size_t)3 * H_ * H_ * 2;
  unsigned short* h_bf   = (unsigned short*)p; p += (size_t)B_ * H_ * 2;
  float*          h_f    = (float*)p;          p += (size_t)B_ * H_ * 4;
  unsigned*       cnt    = (unsigned*)p;

  gru_init_kernel<<<2048, 256, 0, stream>>>(x, W_ih, W_hh, x_bf, wih_bf, whh_bf,
                                            h_bf, h_f, cnt);
  gru_persist_kernel<<<NWG, 256, LDS_BYTES, stream>>>(x_bf, wih_bf, whh_bf,
                                                      b_ih, b_hh, h_bf, h_f, cnt);
  gru_head_kernel<<<1, 256, 0, stream>>>(h_f, headw, headb, y);
}
